// MultiHeadAtten_64364379898184
// MI455X (gfx1250) — matrix operand order repaired and checked by reading, not compiled, not run on hardware
//
#include <hip/hip_runtime.h>
#include <stdint.h>

// ---------- types ----------
typedef __attribute__((ext_vector_type(16))) __bf16         bf16x16;
typedef __attribute__((ext_vector_type(8)))  float          f32x8;
typedef __attribute__((ext_vector_type(4)))  float          f32x4;
typedef __attribute__((ext_vector_type(4)))  unsigned int   u32x4;
typedef __attribute__((ext_vector_type(4)))  unsigned short u16x4;
typedef unsigned short ushort_t;

union Frag  { u32x4 q[2]; bf16x16 v; };
union Pack8 { unsigned short s[8]; u32x4 q; };

__device__ __forceinline__ unsigned short f2bf(float x) {
    unsigned u = __float_as_uint(x);
    unsigned r = u + 0x7fffu + ((u >> 16) & 1u);   // round-to-nearest-even
    return (unsigned short)(r >> 16);
}

// A fragment (16x32, bf16): lane l holds row l%16; K chunks {8h..8h+7, 16+8h..+7}, h=l/16
__device__ __forceinline__ bf16x16 load_frag_A(const ushort_t* base, int ld,
                                               int row, int col, int lane) {
    const int half = lane >> 4, r = lane & 15;
    const ushort_t* p = base + (size_t)(row + r) * ld + col + 8 * half;
    Frag f;
    f.q[0] = *(const u32x4*)p;
    f.q[1] = *(const u32x4*)(p + 16);
    return f.v;
}

// B fragment (32x16, bf16) from row-major "Bt[n][k]": lane n holds col n%16, K chunks {8h..8h+7, 16+8h..+7}
__device__ __forceinline__ bf16x16 load_frag_B(const ushort_t* base, int ld,
                                               int nrow, int kcol, int lane) {
    const int half = lane >> 4, r = lane & 15;
    const ushort_t* p = base + (size_t)(nrow + r) * ld + kcol + 8 * half;
    Frag f;
    f.q[0] = *(const u32x4*)p;
    f.q[1] = *(const u32x4*)(p + 16);
    return f.v;
}

__device__ __forceinline__ f32x8 wmma_bf16(bf16x16 a, bf16x16 b, f32x8 c) {
    return __builtin_amdgcn_wmma_f32_16x16x32_bf16(false, a, false, b,
                                                   (short)0, c, false, false);
}

// ---------- kernel 0a: fp32 -> bf16 row-major ----------
__global__ void cvt_bf16_kernel(const float* __restrict__ in,
                                ushort_t* __restrict__ out, int n4) {
    int i = blockIdx.x * blockDim.x + threadIdx.x;
    if (i >= n4) return;
    f32x4 v = *(const f32x4*)(in + 4 * (size_t)i);
    u16x4 o;
    o.x = f2bf(v.x); o.y = f2bf(v.y); o.z = f2bf(v.z); o.w = f2bf(v.w);
    *(u16x4*)(out + 4 * (size_t)i) = o;
}

// ---------- kernel 0b: fp32 (R x C) -> bf16 transposed (C x R) ----------
__global__ void transpose_cvt_kernel(const float* __restrict__ in,
                                     ushort_t* __restrict__ out, int R, int C) {
    __shared__ float tile[32][33];
    const int tx = threadIdx.x & 31, ty = threadIdx.x >> 5;   // 32x8 threads
    const int bx = blockIdx.x * 32, by = blockIdx.y * 32;
#pragma unroll
    for (int k = 0; k < 32; k += 8) {
        int r = by + ty + k, c = bx + tx;
        if (r < R && c < C) tile[ty + k][tx] = in[(size_t)r * C + c];
    }
    __syncthreads();
#pragma unroll
    for (int k = 0; k < 32; k += 8) {
        int r = bx + ty + k, c = by + tx;                     // out[C][R]
        if (r < C && c < R) out[(size_t)r * R + c] = f2bf(tile[tx][ty + k]);
    }
}

// ---------- kernel 1: C = act(Xb @ Wt^T + bias), bf16 WMMA, 32x64 per wave ----------
// Software-pipelined k-loop (K is a compile-time constant -> full unroll).
// mode 0: bf16 row-major out (M x N), LDS-staged coalesced b128 stores.
// mode 1: bias+LeakyReLU, store transposed per batch of 1024 rows:
//         Vt[batch][f(=N col)][j(=row%1024)], N==128.
template <int K>
__global__ __launch_bounds__(256)
void gemm_bf16_kernel(const ushort_t* __restrict__ A,
                      const ushort_t* __restrict__ Bt,
                      const float* __restrict__ bias,
                      ushort_t* __restrict__ out,
                      int M, int N, int mode) {
    const int lane = threadIdx.x & 31;
    const int w    = threadIdx.x >> 5;
    const int wid  = blockIdx.x * (blockDim.x >> 5) + w;
    const int nT   = N / 64;
    const int mt   = (wid / nT) * 32;
    const int nt   = (wid % nT) * 64;
    if (mt >= M) return;

    const f32x8 zero = {0.f, 0.f, 0.f, 0.f, 0.f, 0.f, 0.f, 0.f};
    f32x8 c[2][4];
#pragma unroll
    for (int mi = 0; mi < 2; ++mi)
#pragma unroll
        for (int ni = 0; ni < 4; ++ni) c[mi][ni] = zero;

    // ---- pipelined main loop: prefetch k+32 while issuing WMMAs for k ----
    bf16x16 a0 = load_frag_A(A, K, mt,      0, lane);
    bf16x16 a1 = load_frag_A(A, K, mt + 16, 0, lane);
    bf16x16 b0 = load_frag_B(Bt, K, nt +  0, 0, lane);
    bf16x16 b1 = load_frag_B(Bt, K, nt + 16, 0, lane);
    bf16x16 b2 = load_frag_B(Bt, K, nt + 32, 0, lane);
    bf16x16 b3 = load_frag_B(Bt, K, nt + 48, 0, lane);

#pragma unroll
    for (int k0 = 32; k0 < K; k0 += 32) {
        bf16x16 na0 = load_frag_A(A, K, mt,      k0, lane);
        bf16x16 na1 = load_frag_A(A, K, mt + 16, k0, lane);
        bf16x16 nb0 = load_frag_B(Bt, K, nt +  0, k0, lane);
        bf16x16 nb1 = load_frag_B(Bt, K, nt + 16, k0, lane);
        bf16x16 nb2 = load_frag_B(Bt, K, nt + 32, k0, lane);
        bf16x16 nb3 = load_frag_B(Bt, K, nt + 48, k0, lane);

        c[0][0] = wmma_bf16(a0, b0, c[0][0]);
        c[1][0] = wmma_bf16(a1, b0, c[1][0]);
        c[0][1] = wmma_bf16(a0, b1, c[0][1]);
        c[1][1] = wmma_bf16(a1, b1, c[1][1]);
        c[0][2] = wmma_bf16(a0, b2, c[0][2]);
        c[1][2] = wmma_bf16(a1, b2, c[1][2]);
        c[0][3] = wmma_bf16(a0, b3, c[0][3]);
        c[1][3] = wmma_bf16(a1, b3, c[1][3]);

        a0 = na0; a1 = na1; b0 = nb0; b1 = nb1; b2 = nb2; b3 = nb3;
    }
    // final step
    c[0][0] = wmma_bf16(a0, b0, c[0][0]);
    c[1][0] = wmma_bf16(a1, b0, c[1][0]);
    c[0][1] = wmma_bf16(a0, b1, c[0][1]);
    c[1][1] = wmma_bf16(a1, b1, c[1][1]);
    c[0][2] = wmma_bf16(a0, b2, c[0][2]);
    c[1][2] = wmma_bf16(a1, b2, c[1][2]);
    c[0][3] = wmma_bf16(a0, b3, c[0][3]);
    c[1][3] = wmma_bf16(a1, b3, c[1][3]);

    const int half = lane >> 4, rr = lane & 15;
    if (mode == 0) {
        // Stage the 32x64 bf16 tile per wave in LDS, then coalesced b128 stores.
        // Row pitch 72 ushorts = 144 B (multiple of 16 B, staggers banks).
        __shared__ __align__(16) ushort_t st[8][32][72];
#pragma unroll
        for (int mi = 0; mi < 2; ++mi)
#pragma unroll
            for (int ni = 0; ni < 4; ++ni) {
                int colL = ni * 16 + rr;
                float bv = bias[nt + colL];
#pragma unroll
                for (int r = 0; r < 8; ++r)
                    st[w][mi * 16 + r + 8 * half][colL] = f2bf(c[mi][ni][r] + bv);
            }
        // same-wave LDS RAW: DS ops are in-order; compiler inserts the wait
#pragma unroll
        for (int cb = 0; cb < 4; ++cb) {
            u32x4 v = *(const u32x4*)&st[w][lane][cb * 16];
            *(u32x4*)(out + (size_t)(mt + lane) * N + nt + cb * 16) = v;
        }
    } else {
#pragma unroll
        for (int mi = 0; mi < 2; ++mi)
#pragma unroll
            for (int ni = 0; ni < 4; ++ni) {
                int col = nt + ni * 16 + rr;          // f index (< 128)
                float bv = bias[col];
                Pack8 pk;
#pragma unroll
                for (int r = 0; r < 8; ++r) {
                    float x = c[mi][ni][r] + bv;
                    x = x > 0.f ? x : 0.01f * x;      // LeakyReLU
                    pk.s[r] = f2bf(x);
                }
                int m = mt + mi * 16 + 8 * half;      // 8 consecutive rows start
                int batch = m >> 10, jb = m & 1023;
                *(u32x4*)(out + ((size_t)batch * 128 + col) * 1024 + jb) = pk.q;
            }
    }
}

// ---------- kernel 2: fused attention; block = (batch, 16-row tile), wave = head ----------
// Per 32-j chunk: 8 score WMMAs (two 16-j subtiles) -> ONE exchange phase
// (3 barriers/chunk instead of 5), head-softmax over 2 positions per thread,
// then 8 P@V WMMAs.
__global__ __launch_bounds__(256)
void attn_kernel(const ushort_t* __restrict__ Qb,
                 const ushort_t* __restrict__ Kb,
                 const ushort_t* __restrict__ Vt,
                 float* __restrict__ O) {
    const int b  = blockIdx.y;
    const int i0 = blockIdx.x * 16;
    const int h    = threadIdx.x >> 5;       // head = wave id (8 waves)
    const int lane = threadIdx.x & 31;
    const int half = lane >> 4, rr = lane & 15;

    __shared__ float    sS[8][16][32];                      // per-head scores (f32), 16 KB
    __shared__ __align__(16) ushort_t sP[8][16][32];        // per-head probs (bf16), 8 KB

    const ushort_t* Qrow = Qb + (size_t)b * 1024 * 1024;
    const ushort_t* Krow = Kb + (size_t)b * 1024 * 1024;
    const ushort_t* Vb   = Vt + (size_t)b * 128  * 1024;    // [f][j]

    // Preload this head's Q A-fragments (16 rows x 128 cols -> 4 k-steps), reused for all j
    bf16x16 aQ[4];
#pragma unroll
    for (int kk = 0; kk < 4; ++kk)
        aQ[kk] = load_frag_A(Qrow, 1024, i0, h * 128 + kk * 32, lane);

    const f32x8 zero = {0.f, 0.f, 0.f, 0.f, 0.f, 0.f, 0.f, 0.f};
    f32x8 hAcc[8];
#pragma unroll
    for (int ft = 0; ft < 8; ++ft) hAcc[ft] = zero;

    const float scale = 0.08838834764831845f;               // 1/sqrt(128)

    for (int j0 = 0; j0 < 1024; j0 += 32) {
        // ---- scores for both 16-j subtiles ----
        f32x8 s0 = zero, s1 = zero;
        {
            bf16x16 bK[4];
#pragma unroll
            for (int kk = 0; kk < 4; ++kk)
                bK[kk] = load_frag_B(Krow, 1024, j0, h * 128 + kk * 32, lane);
#pragma unroll
            for (int kk = 0; kk < 4; ++kk)
                s0 = wmma_bf16(aQ[kk], bK[kk], s0);
#pragma unroll
            for (int kk = 0; kk < 4; ++kk)
                bK[kk] = load_frag_B(Krow, 1024, j0 + 16, h * 128 + kk * 32, lane);
#pragma unroll
            for (int kk = 0; kk < 4; ++kk)
                s1 = wmma_bf16(aQ[kk], bK[kk], s1);
        }
#pragma unroll
        for (int r = 0; r < 8; ++r) {
            sS[h][r + 8 * half][rr]      = s0[r];
            sS[h][r + 8 * half][16 + rr] = s1[r];
        }
        __syncthreads();

        // ---- softmax over HEADS: each thread owns two (i,j) positions ----
        {
            const int t  = threadIdx.x;
            const int ii = t >> 4, jj = t & 15;
#pragma unroll
            for (int p = 0; p < 2; ++p) {
                const int col = jj + p * 16;
                float v[8];
                float m = -3.402823466e38f;
#pragma unroll
                for (int hh = 0; hh < 8; ++hh) {
                    v[hh] = sS[hh][ii][col] * scale;
                    m = fmaxf(m, v[hh]);
                }
                float sum = 0.f;
#pragma unroll
                for (int hh = 0; hh < 8; ++hh) {
                    v[hh] = __expf(v[hh] - m);
                    sum += v[hh];
                }
                float inv = 1.0f / sum;
#pragma unroll
                for (int hh = 0; hh < 8; ++hh)
                    sP[hh][ii][col] = f2bf(v[hh] * inv);
            }
        }
        __syncthreads();

        // ---- H_h += P_h (16x32) @ V (32x128); A-fragment of P from LDS (ld = 32) ----
        bf16x16 aP;
        {
            const ushort_t* p = &sP[h][rr][0] + 8 * half;
            Frag f;
            f.q[0] = *(const u32x4*)p;
            f.q[1] = *(const u32x4*)(p + 16);
            aP = f.v;
        }
#pragma unroll
        for (int g = 0; g < 2; ++g) {
            bf16x16 bV[4];
#pragma unroll
            for (int ft = 0; ft < 4; ++ft)
                bV[ft] = load_frag_B(Vb, 1024, (g * 4 + ft) * 16, j0, lane);
#pragma unroll
            for (int ft = 0; ft < 4; ++ft)
                hAcc[g * 4 + ft] = wmma_bf16(aP, bV[ft], hAcc[g * 4 + ft]);
        }
        __syncthreads();   // protect sP before next chunk overwrites it
    }

    // O[b][i][h*128 + f]  (fp32)
#pragma unroll
    for (int ft = 0; ft < 8; ++ft)
#pragma unroll
        for (int r = 0; r < 8; ++r) {
            int row = i0 + r + 8 * half;
            O[((size_t)b * 1024 + row) * 1024 + h * 128 + ft * 16 + rr] = hAcc[ft][r];
        }
}

// ---------- host launcher ----------
extern "C" void kernel_launch(void* const* d_in, const int* in_sizes, int n_in,
                              void* d_out, int out_size, void* d_ws, size_t ws_size,
                              hipStream_t stream) {
    (void)in_sizes; (void)n_in; (void)out_size; (void)ws_size;
    const float* X  = (const float*)d_in[0];
    const float* Wq = (const float*)d_in[1];
    const float* bq = (const float*)d_in[2];
    const float* Wk = (const float*)d_in[3];
    const float* bk = (const float*)d_in[4];
    const float* Wv = (const float*)d_in[5];
    const float* bv = (const float*)d_in[6];
    float* O = (float*)d_out;

    // workspace layout (bytes)
    char* ws = (char*)d_ws;
    size_t o = 0;
    ushort_t* Xb  = (ushort_t*)(ws + o); o += (size_t)8192 * 1024 * 2;   // X bf16
    ushort_t* Wqt = (ushort_t*)(ws + o); o += (size_t)1024 * 1024 * 2;   // Wq^T bf16
    ushort_t* Wkt = (ushort_t*)(ws + o); o += (size_t)1024 * 1024 * 2;   // Wk^T bf16
    ushort_t* Wvt = (ushort_t*)(ws + o); o += (size_t)128  * 1024 * 2;   // Wv^T bf16
    ushort_t* Qb  = (ushort_t*)(ws + o); o += (size_t)8192 * 1024 * 2;   // Q bf16
    ushort_t* Kb  = (ushort_t*)(ws + o); o += (size_t)8192 * 1024 * 2;   // K bf16
    ushort_t* Vt  = (ushort_t*)(ws + o); o += (size_t)8 * 128 * 1024 * 2;// V^T bf16

    // 0) conversions / transposes
    {
        int n4 = (8192 * 1024) / 4;
        cvt_bf16_kernel<<<(n4 + 255) / 256, 256, 0, stream>>>(X, Xb, n4);
        transpose_cvt_kernel<<<dim3(32, 32), 256, 0, stream>>>(Wq, Wqt, 1024, 1024);
        transpose_cvt_kernel<<<dim3(32, 32), 256, 0, stream>>>(Wk, Wkt, 1024, 1024);
        transpose_cvt_kernel<<<dim3(4,  32), 256, 0, stream>>>(Wv, Wvt, 1024, 128);
    }

    // 1) projections (bf16 WMMA GEMMs), K = 1024 compile-time
    {
        // Q, K: M=8192, N=1024 -> (256 * 16) waves / 8 waves-per-block
        gemm_bf16_kernel<1024><<<512, 256, 0, stream>>>(Xb, Wqt, bq, Qb, 8192, 1024, 0);
        gemm_bf16_kernel<1024><<<512, 256, 0, stream>>>(Xb, Wkt, bk, Kb, 8192, 1024, 0);
        // V: M=8192, N=128 -> (256 * 2) waves / 8
        gemm_bf16_kernel<1024><<<64, 256, 0, stream>>>(Xb, Wvt, bv, Vt, 8192, 128, 1);
    }

    // 2) fused scores + head-softmax + P@V
    attn_kernel<<<dim3(64, 8), 256, 0, stream>>>(Qb, Kb, Vt, O);
}
